// MTTT_22110491640050
// MI455X (gfx1250) — matrix operand order repaired and checked by reading, not compiled, not run on hardware
//
#include <hip/hip_runtime.h>
#include <hip/hip_bf16.h>

#define NH   6
#define DIM  768
#define HD   128
#define TSEQ 2048
#define BB   2
#define BT   (BB * TSEQ)      // 4096
#define EPSL 1e-6f

typedef __attribute__((ext_vector_type(16))) __bf16 v16bf;
typedef __attribute__((ext_vector_type(8)))  __bf16 v8bf;
typedef __attribute__((ext_vector_type(8)))  float  v8f;

// ---------- helpers ----------
__device__ __forceinline__ unsigned short f2bf(float f) {
    unsigned int u = __float_as_uint(f);
    unsigned int r = (u + 0x7FFFu + ((u >> 16) & 1u)) >> 16;   // RNE
    return (unsigned short)r;
}
__device__ __forceinline__ float bf2f(unsigned short b) {
    return __uint_as_float(((unsigned int)b) << 16);
}
__device__ __forceinline__ float wave_sum(float v) {
#pragma unroll
    for (int m = 16; m >= 1; m >>= 1) v += __shfl_xor(v, m, 32);
    return v;
}
// block of 768 threads (24 wave32); red must have >= 32 floats
__device__ __forceinline__ float block_sum(float v, float* red, int tid) {
    v = wave_sum(v);
    if ((tid & 31) == 0) red[tid >> 5] = v;
    __syncthreads();
    if (tid < 32) {
        float x = (tid < 24) ? red[tid] : 0.f;
        x = wave_sum(x);
        if (tid == 0) red[31] = x;
    }
    __syncthreads();
    float r = red[31];
    __syncthreads();
    return r;
}

// ---------- prep: fp32 -> hi/lo bf16 ----------
__global__ void mttt_prep_x(const float* __restrict__ x,
                            unsigned short* __restrict__ xhi,
                            unsigned short* __restrict__ xlo, int n) {
    for (int i = blockIdx.x * blockDim.x + threadIdx.x; i < n;
         i += gridDim.x * blockDim.x) {
        float v = x[i];
        unsigned short h = f2bf(v);
        xhi[i] = h;
        xlo[i] = f2bf(v - bf2f(h));
    }
}

// phi/psi [h][D][HD] -> transposed [proj][h][HD][D] hi/lo bf16
__global__ void mttt_prep_w(const float* __restrict__ phi,
                            const float* __restrict__ psi,
                            unsigned short* __restrict__ whi,
                            unsigned short* __restrict__ wlo) {
    const int n = 2 * NH * HD * DIM;
    for (int idx = blockIdx.x * blockDim.x + threadIdx.x; idx < n;
         idx += gridDim.x * blockDim.x) {
        int proj = idx / (NH * HD * DIM);
        int rem  = idx % (NH * HD * DIM);
        int hh   = rem / (HD * DIM);
        int r2   = rem % (HD * DIM);
        int nn   = r2 / DIM;
        int kk   = r2 % DIM;
        const float* src = proj ? psi : phi;
        float v = src[((size_t)hh * DIM + kk) * HD + nn];
        unsigned short h = f2bf(v);
        whi[idx] = h;
        wlo[idx] = f2bf(v - bf2f(h));
    }
}

__global__ void mttt_zero(float* __restrict__ p, int n) {
    for (int i = blockIdx.x * blockDim.x + threadIdx.x; i < n;
         i += gridDim.x * blockDim.x) p[i] = 0.f;
}

// ---------- WMMA projection: Z[proj][h][bt][hd] = X @ {phi,psi}[h] ----------
__global__ void __launch_bounds__(256)
mttt_proj_wmma(const unsigned short* __restrict__ xhi,
               const unsigned short* __restrict__ xlo,
               const unsigned short* __restrict__ whi,
               const unsigned short* __restrict__ wlo,
               float* __restrict__ Z) {
    const int wave = (int)(threadIdx.x >> 5);
    const int lane = (int)(threadIdx.x & 31);
    const int tilesPerMat = (BT / 16) * (HD / 16);   // 2048
    int tile = blockIdx.x * 8 + wave;
    if (tile >= 12 * tilesPerMat) return;
    int mat = tile / tilesPerMat;                    // proj*NH + h
    int r   = tile % tilesPerMat;
    int mt = r >> 3, nt = r & 7;
    int hi2 = lane >> 4, ln = lane & 15;
    const int m0 = mt * 16, n0 = nt * 16;

    const unsigned short* arh = xhi + (size_t)(m0 + ln) * DIM;
    const unsigned short* arl = xlo + (size_t)(m0 + ln) * DIM;
    const unsigned short* brh = whi + (size_t)mat * HD * DIM + (size_t)(n0 + ln) * DIM;
    const unsigned short* brl = wlo + (size_t)mat * HD * DIM + (size_t)(n0 + ln) * DIM;

    v8f c = {};
    for (int k0 = 0; k0 < DIM; k0 += 32) {
        // A fragment: rows per lane, K = k0+hi2*16 .. +15 (contiguous)
        v8bf a0h = *(const v8bf*)(arh + k0 + hi2 * 16);
        v8bf a1h = *(const v8bf*)(arh + k0 + hi2 * 16 + 8);
        v8bf a0l = *(const v8bf*)(arl + k0 + hi2 * 16);
        v8bf a1l = *(const v8bf*)(arl + k0 + hi2 * 16 + 8);
        v16bf ah, al;
#pragma unroll
        for (int i = 0; i < 8; ++i) {
            ah[i] = a0h[i]; ah[8 + i] = a1h[i];
            al[i] = a0l[i]; al[8 + i] = a1l[i];
        }
        // B fragment: col per lane, K = k0 + hi2*16 .. +15 (contiguous in wT)
        v16bf bh = *(const v16bf*)(brh + k0 + hi2 * 16);
        v16bf bl = *(const v16bf*)(brl + k0 + hi2 * 16);
        c = __builtin_amdgcn_wmma_f32_16x16x32_bf16(false, ah, false, bh, (short)0, c, false, false);
        c = __builtin_amdgcn_wmma_f32_16x16x32_bf16(false, ah, false, bl, (short)0, c, false, false);
        c = __builtin_amdgcn_wmma_f32_16x16x32_bf16(false, al, false, bh, (short)0, c, false, false);
    }
    float* zb = Z + (size_t)mat * BT * HD;
#pragma unroll
    for (int rr = 0; rr < 8; ++rr) {
        int m = m0 + hi2 * 8 + rr;
        zb[(size_t)m * HD + n0 + ln] = c[rr];
    }
}

// ---------- sequential TTT scan: one block per (b, head) ----------
__global__ void __launch_bounds__(768, 1)
mttt_scan(const float* __restrict__ x,     // [B][T][D]
          const float* __restrict__ W0,    // [NH][HD][HD]
          const float* __restrict__ g,     // [NH][HD][D]
          const float* __restrict__ h,     // [NH][HD][D]
          const float* __restrict__ lns,   // [NH][D]
          const float* __restrict__ lnb,   // [NH][D]
          const float* __restrict__ Z,     // [2][NH][BT][HD]
          float* __restrict__ out) {       // [B][T][D]
    extern __shared__ float smem[];
    float* Ws  = smem;                 // 16384
    float* zb  = Ws + HD * HD;         // 128
    float* z2b = zb + HD;              // 128
    float* yb  = z2b + HD;             // 128
    float* y2b = yb + HD;              // 128
    float* dyb = y2b + HD;             // 128
    float* ob  = dyb + HD;             // 128
    float* dpb = ob + HD;              // 768
    float* red = dpb + DIM;            // 32

    const int tid  = (int)threadIdx.x;
    const int bh   = (int)blockIdx.x;  // 0..11
    const int b    = bh / NH;
    const int head = bh % NH;

    const float* W0h  = W0 + (size_t)head * HD * HD;
    const float* gh   = g  + (size_t)head * HD * DIM;
    const float* hh   = h  + (size_t)head * HD * DIM;
    const float* zphi = Z + ((size_t)head * BT + (size_t)b * TSEQ) * HD;
    const float* zpsi = Z + ((size_t)(NH + head) * BT + (size_t)b * TSEQ) * HD;
    const float  sd = lns[(size_t)head * DIM + tid];
    const float  bd = lnb[(size_t)head * DIM + tid];

    for (int i = tid; i < HD * HD; i += 768) Ws[i] = W0h[i];
    __syncthreads();

    const int wid = tid >> 5, lane = tid & 31;

    for (int t = 0; t < TSEQ; ++t) {
        if (tid < HD) {
            zb[tid]  = zphi[(size_t)t * HD + tid];
            z2b[tid] = zpsi[(size_t)t * HD + tid];
        }
        __syncthreads();
        // y = z@W, y2 = z2@W (old W), per output column from LDS
        if (tid < HD) {
            float a = 0.f, a2 = 0.f;
            for (int i = 0; i < HD; ++i) {
                float w = Ws[i * HD + tid];
                a += zb[i] * w; a2 += z2b[i] * w;
            }
            yb[tid] = a; y2b[tid] = a2;
        }
        __syncthreads();
        // p_d = sum_j y[j]*g[j][d]  (coalesced over d = tid)
        float p = 0.f;
        {
            const float* gc = gh + tid;
            for (int j = 0; j < HD; ++j) p += yb[j] * gc[(size_t)j * DIM];
        }
        float mu   = block_sum(p, red, tid) * (1.f / DIM);
        float sp2  = block_sum(p * p, red, tid) * (1.f / DIM);
        float var  = sp2 - mu * mu;
        float rstd = rsqrtf(var + EPSL);
        float xh   = (p - mu) * rstd;
        float o_   = xh * sd + bd;
        float xt   = x[((size_t)b * TSEQ + t) * DIM + tid];
        float dout = (o_ - xt) * ((float)NH / (float)DIM);
        float dxh  = dout * sd;
        float s1   = block_sum(dxh, red, tid) * (1.f / DIM);
        float s2   = block_sum(dxh * xh, red, tid) * (1.f / DIM);
        float dp   = rstd * (dxh - s1 - xh * s2);
        dpb[tid] = dp;
        // c = z2 . z (barriers inside also publish dpb)
        float cz = block_sum(tid < HD ? zb[tid] * z2b[tid] : 0.f, red, tid);
        // dy[j] = g[j][:] . dp  — 16 waves x 8 rows, lane-strided coalesced rows
        if (wid < 16) {
            for (int rr = 0; rr < 8; ++rr) {
                int j = wid * 8 + rr;
                const float* grow = gh + (size_t)j * DIM;
                float acc = 0.f;
                for (int dd = lane; dd < DIM; dd += 32) acc += grow[dd] * dpb[dd];
                acc = wave_sum(acc);
                if (lane == 0) dyb[j] = acc;
            }
        }
        __syncthreads();
        // o = z2@Wnew = y2 - (z2.z)*dy ; update W in LDS: W -= z (x) dy
        if (tid < HD) ob[tid] = y2b[tid] - cz * dyb[tid];
        for (int idx = tid; idx < HD * HD; idx += 768) {
            int i = idx >> 7, j = idx & (HD - 1);
            Ws[idx] -= zb[i] * dyb[j];
        }
        __syncthreads();
        // out_d += sum_j o[j]*h[j][d]
        float acc = 0.f;
        const float* hc = hh + tid;
        for (int j = 0; j < HD; ++j) acc += ob[j] * hc[(size_t)j * DIM];
        unsafeAtomicAdd(&out[((size_t)b * TSEQ + t) * DIM + tid], acc);
        __syncthreads();
    }
}

// ---------- host ----------
extern "C" void kernel_launch(void* const* d_in, const int* in_sizes, int n_in,
                              void* d_out, int out_size, void* d_ws, size_t ws_size,
                              hipStream_t stream) {
    const float* batch = (const float*)d_in[0];
    const float* psi   = (const float*)d_in[1];
    const float* phi   = (const float*)d_in[2];
    const float* W0    = (const float*)d_in[3];
    const float* g     = (const float*)d_in[4];
    const float* h     = (const float*)d_in[5];
    const float* lns   = (const float*)d_in[6];
    const float* lnb   = (const float*)d_in[7];
    float* out = (float*)d_out;

    char* ws = (char*)d_ws;
    const size_t SZ_X  = (size_t)BT * DIM * sizeof(unsigned short);       // 6.29 MB
    const size_t SZ_WT = (size_t)2 * NH * HD * DIM * sizeof(unsigned short); // 2.36 MB
    unsigned short* xhi = (unsigned short*)(ws);
    unsigned short* xlo = (unsigned short*)(ws + SZ_X);
    unsigned short* whi = (unsigned short*)(ws + 2 * SZ_X);
    unsigned short* wlo = (unsigned short*)(ws + 2 * SZ_X + SZ_WT);
    float* Z = (float*)(ws + 2 * SZ_X + 2 * SZ_WT);                        // 25.2 MB f32

    const int nX = BT * DIM;
    mttt_prep_x<<<2048, 256, 0, stream>>>(batch, xhi, xlo, nX);
    mttt_prep_w<<<2048, 256, 0, stream>>>(phi, psi, whi, wlo);
    mttt_zero<<<2048, 256, 0, stream>>>(out, BB * TSEQ * DIM);

    // 12 mats x 2048 tiles, 8 tiles (waves) per 256-thread block
    mttt_proj_wmma<<<(12 * 2048) / 8, 256, 0, stream>>>(xhi, xlo, whi, wlo, Z);

    const size_t shm = (size_t)(HD * HD + 6 * HD + DIM + 32) * sizeof(float); // ~71.8 KB
    mttt_scan<<<BB * NH, 768, shm, stream>>>(batch, W0, g, h, lns, lnb, Z, out);
}
